// SRP_53085795778797
// MI455X (gfx1250) — matrix operand order repaired and checked by reading, not compiled, not run on hardware
//
#include <hip/hip_runtime.h>
#include <hip/hip_bf16.h>
#include <math.h>

// ---------------- problem constants (match reference) ----------------
#define BATCH      16
#define N_MICS     6
#define N_SAMPLES  64000
#define WIN        400
#define HOP        160
#define NFFT       512
#define FBINS      256
#define T_FRAMES   398            // (64000-400)/160 + 1
#define N_PAIRS    15
#define D_DIRS     2701           // 73 * 37
#define N_EL       37
#define K_TOT      7680           // 2 * 15 * 256
#define KT         240            // K_TOT / 32  (even -> clean 2-deep pipeline)
#define MT         398            // (BATCH*T_FRAMES) / 16  (exact, even)
#define NTP        176            // padded N tiles of 16 (169 real -> 176 = 11 blocks * 16)
#define TWO_PI     6.28318530717958647692f

typedef _Float16 v16h __attribute__((ext_vector_type(16)));
typedef float    v8f  __attribute__((ext_vector_type(8)));

__device__ __constant__ int c_MI[N_PAIRS] = {0,0,0,0,0,1,1,1,1,2,2,2,3,3,4};
__device__ __constant__ int c_MJ[N_PAIRS] = {1,2,3,4,5,2,3,4,5,3,4,5,4,5,5};

// =====================================================================
// Kernel 1: windowed frames -> 256-bin DFT (6 mics) -> PHAT cross-power
//           spectra, stored straight into WMMA A-fragment layout (f16).
// One block per output row r = b*T + t.  256 threads = one freq bin each.
// Frames stored with stride 8 (32 B) so each DFT tap is 2x ds_load_b128.
// =====================================================================
__global__ __launch_bounds__(256) void srp_cps_kernel(const float* __restrict__ x,
                                                      _Float16* __restrict__ Apk) {
    __shared__ float sm[WIN * 8];               // [n][0..5 mics, 2 pad], 12.8 KB
    const int r   = blockIdx.x;                 // b*T + t
    const int b   = r / T_FRAMES;
    const int t   = r % T_FRAMES;
    const int tid = threadIdx.x;

    // cooperative load: coalesced over n, window applied on the fly
    for (int m = 0; m < N_MICS; ++m) {
        const float* src = x + ((size_t)b * N_MICS + m) * N_SAMPLES + (size_t)t * HOP;
        for (int n = tid; n < WIN; n += 256) {
            float w = 0.5f - 0.5f * __cosf(TWO_PI * (float)n / (float)WIN);
            sm[n * 8 + m] = src[n] * w;
        }
    }
    __syncthreads();

    // direct DFT, bin k = tid: X_k = sum_n xw[n] * e^{-i 2pi k n / 512}
    const int k = tid;
    float cs, ss;
    __sincosf(TWO_PI * (float)k / (float)NFFT, &ss, &cs);
    float c = 1.0f, d = 0.0f;                   // e^{-i*0}; d tracks -sin
    float re[N_MICS] = {0.f,0.f,0.f,0.f,0.f,0.f};
    float im[N_MICS] = {0.f,0.f,0.f,0.f,0.f,0.f};
    const float4* smv = reinterpret_cast<const float4*>(sm);
    for (int n = 0; n < WIN; ++n) {
        const float4 x03 = smv[n * 2 + 0];      // mics 0..3 (uniform addr -> broadcast)
        const float4 x45 = smv[n * 2 + 1];      // mics 4,5 (+pad)
        re[0] = fmaf(x03.x, c, re[0]); im[0] = fmaf(x03.x, d, im[0]);
        re[1] = fmaf(x03.y, c, re[1]); im[1] = fmaf(x03.y, d, im[1]);
        re[2] = fmaf(x03.z, c, re[2]); im[2] = fmaf(x03.z, d, im[2]);
        re[3] = fmaf(x03.w, c, re[3]); im[3] = fmaf(x03.w, d, im[3]);
        re[4] = fmaf(x45.x, c, re[4]); im[4] = fmaf(x45.x, d, im[4]);
        re[5] = fmaf(x45.y, c, re[5]); im[5] = fmaf(x45.y, d, im[5]);
        // rotate (c + i d) by e^{-i 2pi k/512} = (cs - i ss)
        float c2 = fmaf(c, cs,  d * ss);
        float d2 = fmaf(d, cs, -c * ss);
        c = c2; d = d2;
    }

    // scatter into WMMA A-fragment layout (16-bit A 16x32, contiguous k = 16*(L/16)+h)
    const int mt  = r >> 4;
    const int m16 = r & 15;
    #pragma unroll
    for (int p = 0; p < N_PAIRS; ++p) {
        const int i = c_MI[p], j = c_MJ[p];
        // cps = S_j * conj(S_i), PHAT normalized
        float rc  = re[j] * re[i] + im[j] * im[i];
        float ic  = im[j] * re[i] - re[j] * im[i];
        float inv = 1.0f / (sqrtf(rc * rc + ic * ic) + 1e-8f);
        rc *= inv; ic *= inv;
        #pragma unroll
        for (int part = 0; part < 2; ++part) {
            const int   kk = part * 3840 + p * FBINS + k;
            const float v  = part ? ic : rc;
            const int   kt = kk >> 5, ko = kk & 31, g = ko >> 3;
            const int   L  = m16 + ((ko >> 4) << 4);
            const int   h  = ko & 15;
            Apk[(((size_t)mt * KT + kt) * 32 + L) * 16 + h] = (_Float16)v;
        }
    }
}

// =====================================================================
// Kernel 2: steering phase shifts B = [Re(ps); -Im(ps)] generated
//           directly in WMMA B-fragment layout (32x16 f16: lane L holds
//           K = 16*(L/16)+h for column N = L%16). Zero-padded to NTP.
// One thread per (kt, nt, lane): writes 16 contiguous halves (32 B).
// =====================================================================
__global__ __launch_bounds__(256) void srp_psv_kernel(const float* __restrict__ mic,
                                                      _Float16* __restrict__ Bpk) {
    const long g = (long)blockIdx.x * blockDim.x + threadIdx.x;
    if (g >= (long)KT * NTP * 32) return;
    const int lane = (int)(g & 31);
    const int nt   = (int)((g >> 5) % NTP);
    const int kt   = (int)(g / (32 * NTP));
    _Float16* out  = Bpk + (size_t)g * 16;      // ((kt*NTP+nt)*32+lane)*16

    const int n = nt * 16 + (lane & 15);        // direction index (column)
    if (n >= D_DIRS) {
        #pragma unroll
        for (int h = 0; h < 16; ++h) out[h] = (_Float16)0.0f;
        return;
    }
    // steering vector for direction n = az*37 + el
    const int az = n / N_EL, el = n % N_EL;
    const float deg2rad = TWO_PI / 360.0f;
    const float theta = (float)el * 5.0f * deg2rad;
    const float phi   = ((float)az * 5.0f - 180.0f) * deg2rad;
    float st, ct, sp, cp;
    __sincosf(theta, &st, &ct);
    __sincosf(phi,   &sp, &cp);
    const float sv0 = st * cp, sv1 = st * sp, sv2 = ct;

    const int kbase = kt * 32 + ((lane >> 4) << 4);  // first K of this lane's 16
    const int isIm  = (kbase >= 3840);
    const int kk    = isIm ? kbase - 3840 : kbase;
    const int p     = kk >> 8;                  // constant across the 16 halves
    const int f0    = kk & 255;
    const int i = c_MI[p], j = c_MJ[p];
    const float dx = mic[i * 3 + 0] - mic[j * 3 + 0];
    const float dy = mic[i * 3 + 1] - mic[j * 3 + 1];
    const float dz = mic[i * 3 + 2] - mic[j * 3 + 2];
    const float dtau  = (dx * sv0 + dy * sv1 + dz * sv2) * (1.0f / 343.0f);
    const float wstep = TWO_PI * 31.25f * dtau;      // omega_f * dtau = wstep * f
    #pragma unroll
    for (int h = 0; h < 16; ++h) {
        float s, cc;
        __sincosf(wstep * (float)(f0 + h), &s, &cc);
        out[h] = (_Float16)(isIm ? -s : cc);    // [Re ; -Im] sign fold
    }
}

// =====================================================================
// Kernel 3: C[6368 x 2701] = A[6368 x 7680] * B[7680 x 2701]
// f16 WMMA 16x16x32, fp32 accumulate. 4 waves/block, wave = 32x64 tile
// (2 A frags + 4 B frags -> 8 WMMAs per K-step: halves B-side L2 reads
// vs the 16x64 version). Ping-pong double buffering over K keeps one
// fragment set in flight behind the matrix ops (partial s_wait_loadcnt).
// =====================================================================
__global__ __launch_bounds__(128) void srp_gemm_kernel(const _Float16* __restrict__ Apk,
                                                       const _Float16* __restrict__ Bpk,
                                                       float* __restrict__ out) {
    const int lane = threadIdx.x & 31;
    const int wave = threadIdx.x >> 5;
    const int mt0  = blockIdx.x * 2;                   // two m-tiles per block row
    const int mt1  = mt0 + 1;
    const int ntb  = blockIdx.y * 16 + wave * 4;       // base n-tile (4 tiles/wave)

    v8f acc[2][4] = {};
    // base fragment pointers in v16h (32-byte) units
    const v16h* aptr0 = reinterpret_cast<const v16h*>(Apk) + ((size_t)mt0 * KT * 32 + lane);
    const v16h* aptr1 = reinterpret_cast<const v16h*>(Apk) + ((size_t)mt1 * KT * 32 + lane);
    const v16h* bptr  = reinterpret_cast<const v16h*>(Bpk) + ((size_t)ntb * 32 + lane);

    struct Frags { v16h a0, a1, b0, b1, b2, b3; };
    auto loadf = [&](int kt) {
        Frags f;
        f.a0 = aptr0[(size_t)kt * 32];
        f.a1 = aptr1[(size_t)kt * 32];
        const v16h* bp = bptr + (size_t)kt * (NTP * 32);
        f.b0 = bp[0]; f.b1 = bp[32]; f.b2 = bp[64]; f.b3 = bp[96];
        return f;
    };
    auto mma = [&](const Frags& f) {
        acc[0][0] = __builtin_amdgcn_wmma_f32_16x16x32_f16(false, f.a0, false, f.b0,
                                                           (short)0, acc[0][0], false, false);
        acc[1][0] = __builtin_amdgcn_wmma_f32_16x16x32_f16(false, f.a1, false, f.b0,
                                                           (short)0, acc[1][0], false, false);
        acc[0][1] = __builtin_amdgcn_wmma_f32_16x16x32_f16(false, f.a0, false, f.b1,
                                                           (short)0, acc[0][1], false, false);
        acc[1][1] = __builtin_amdgcn_wmma_f32_16x16x32_f16(false, f.a1, false, f.b1,
                                                           (short)0, acc[1][1], false, false);
        acc[0][2] = __builtin_amdgcn_wmma_f32_16x16x32_f16(false, f.a0, false, f.b2,
                                                           (short)0, acc[0][2], false, false);
        acc[1][2] = __builtin_amdgcn_wmma_f32_16x16x32_f16(false, f.a1, false, f.b2,
                                                           (short)0, acc[1][2], false, false);
        acc[0][3] = __builtin_amdgcn_wmma_f32_16x16x32_f16(false, f.a0, false, f.b3,
                                                           (short)0, acc[0][3], false, false);
        acc[1][3] = __builtin_amdgcn_wmma_f32_16x16x32_f16(false, f.a1, false, f.b3,
                                                           (short)0, acc[1][3], false, false);
    };

    Frags fA = loadf(0);
    for (int kt = 0; kt < KT; kt += 2) {
        Frags fB = loadf(kt + 1);                       // in flight during mma(fA)
        __builtin_prefetch((const void*)(aptr0 + (size_t)(kt + 2) * 32), 0, 1);
        __builtin_prefetch((const void*)(bptr + (size_t)(kt + 2) * (NTP * 32)), 0, 1);
        mma(fA);
        if (kt + 2 < KT) fA = loadf(kt + 2);            // in flight during mma(fB)
        mma(fB);
    }

    // C/D 32-bit layout: VGPR v -> M = v + 8*(lane/16), N = lane%16
    const int mrow = (lane >> 4) * 8;
    const int ncol = lane & 15;
    #pragma unroll
    for (int w = 0; w < 2; ++w) {
        #pragma unroll
        for (int jn = 0; jn < 4; ++jn) {
            const int n = (ntb + jn) * 16 + ncol;
            if (n < D_DIRS) {
                #pragma unroll
                for (int v = 0; v < 8; ++v) {
                    out[(size_t)((mt0 + w) * 16 + mrow + v) * D_DIRS + n] = acc[w][jn][v];
                }
            }
        }
    }
}

// =====================================================================
extern "C" void kernel_launch(void* const* d_in, const int* in_sizes, int n_in,
                              void* d_out, int out_size, void* d_ws, size_t ws_size,
                              hipStream_t stream) {
    const float* x   = (const float*)d_in[0];   // (16, 6, 64000) f32
    const float* mic = (const float*)d_in[1];   // (6, 3) f32
    float*       out = (float*)d_out;           // (16, 398, 73, 37) f32

    _Float16* Apk = (_Float16*)d_ws;                     // 398*240*512 halves = 97.8 MB
    const size_t aElems = (size_t)MT * KT * 512;
    _Float16* Bpk = Apk + aElems;                        // 240*176*512 halves = 43.3 MB

    // 1) STFT + PHAT cross-power spectra -> A fragments
    srp_cps_kernel<<<BATCH * T_FRAMES, 256, 0, stream>>>(x, Apk);

    // 2) steering phase shifts -> B fragments (zero-padded columns)
    const long totB = (long)KT * NTP * 32;               // 1,351,680 threads
    srp_psv_kernel<<<(int)((totB + 255) / 256), 256, 0, stream>>>(mic, Bpk);

    // 3) WMMA GEMM -> output (32-row x 256-col blocks)
    dim3 grid(MT / 2, NTP / 16);                         // 199 x 11 blocks, 128 thr
    srp_gemm_kernel<<<grid, 128, 0, stream>>>(Apk, Bpk, out);
}
